// TextVisualModel_70927089926536
// MI455X (gfx1250) — matrix operand order repaired and checked by reading, not compiled, not run on hardware
//
#include <hip/hip_runtime.h>
#include <stdint.h>

#define BB   64
#define NT   128
#define NV   256
#define DD   512
#define EPSF 1e-7f

typedef _Float16 v16h __attribute__((ext_vector_type(16)));
typedef float    v8f  __attribute__((ext_vector_type(8)));
typedef float    f4   __attribute__((ext_vector_type(4)));
typedef unsigned short u16x4 __attribute__((ext_vector_type(4)));

union Frag { v16h h; f4 f[2]; };
union HCvt { _Float16 h; unsigned short u; };

// ---------------- Kernel 1: L2-normalize rows, convert to f16 ----------------
__global__ __launch_bounds__(256) void normalize_f16_kernel(
    const float* __restrict__ text, const float* __restrict__ vis,
    unsigned short* __restrict__ tn, unsigned short* __restrict__ vn)
{
    const int lane = threadIdx.x & 31;
    const int wave = threadIdx.x >> 5;
    const int row  = blockIdx.x * 8 + wave;
    const int TROWS = BB * NT; // 8192

    const float* src;
    unsigned short* dst;
    if (row < TROWS) { src = text + (size_t)row * DD; dst = tn + (size_t)row * DD; }
    else             { int r = row - TROWS; src = vis + (size_t)r * DD; dst = vn + (size_t)r * DD; }

    const f4* s4 = (const f4*)src;
    f4 vals[4];
    float ss = 0.f;
    #pragma unroll
    for (int i = 0; i < 4; ++i) {
        vals[i] = s4[lane + 32 * i];
        #pragma unroll
        for (int k = 0; k < 4; ++k) ss += vals[i][k] * vals[i][k];
    }
    #pragma unroll
    for (int d = 1; d < 32; d <<= 1) ss += __shfl_xor(ss, d, 32);
    const float inv = 1.0f / fmaxf(sqrtf(ss), 1e-12f);

    u16x4* d4 = (u16x4*)dst;
    #pragma unroll
    for (int i = 0; i < 4; ++i) {
        u16x4 o;
        #pragma unroll
        for (int k = 0; k < 4; ++k) {
            HCvt c; c.h = (_Float16)(vals[i][k] * inv);
            o[k] = c.u;
        }
        d4[lane + 32 * i] = o;
    }
}

// ---------------- Kernel 2: pairwise GEMM + fused max / masked-sum / reg ----------------
#define LDS_ROW_U16 40   // 80-byte padded LDS row stride (64B data + 16B pad)

// Stage one thread's 64B V row slab into LDS, direct global->LDS async copy.
// IOFFSET applies to BOTH the global and LDS address (ISA async pseudocode).
__device__ __forceinline__ void stage_async(uint32_t lds_addr, const void* gaddr) {
    asm volatile(
        "global_load_async_to_lds_b128 %0, %1, off\n\t"
        "global_load_async_to_lds_b128 %0, %1, off offset:16\n\t"
        "global_load_async_to_lds_b128 %0, %1, off offset:32\n\t"
        "global_load_async_to_lds_b128 %0, %1, off offset:48"
        :: "v"(lds_addr), "v"(gaddr) : "memory");
}

__device__ __forceinline__ void loadB(Frag& b, const unsigned short* lv,
                                      int t, int lane_lo, int sel) {
    const f4* p = (const f4*)(lv + (t * 16 + lane_lo) * LDS_ROW_U16 + sel * 16);
    b.f[0] = p[0];
    b.f[1] = p[1];
}

#define SG_DSREAD 0x100
#define SG_WMMA   0x008

__global__ __launch_bounds__(256, 1) void pair_gemm_kernel(
    const unsigned short* __restrict__ tn,
    const unsigned short* __restrict__ vn,
    const int* __restrict__ amask,
    float* __restrict__ msum,   // (64*64) masked row-max sums
    float* __restrict__ regp)   // (64*64) clip^2 partial sums
{
    __shared__ unsigned short ldsV[2][NV * LDS_ROW_U16]; // 2 x 20 KB, double buffered
    __shared__ float ldsRed[16];

    const int i = blockIdx.x;
    const int j = blockIdx.y;
    const int tid = threadIdx.x;
    const int lane = tid & 31;
    const int wave = tid >> 5;
    const int lane_lo = lane & 15;
    const int sel = lane >> 4;           // 0: lanes 0-15, 1: lanes 16-31

    const unsigned short* Tbase = tn + (size_t)i * NT * DD;
    const unsigned short* Vbase = vn + (size_t)j * NV * DD;

    v8f zero = {0.f, 0.f, 0.f, 0.f, 0.f, 0.f, 0.f, 0.f};
    v8f acc[16];
    #pragma unroll
    for (int t = 0; t < 16; ++t) acc[t] = zero;

    const int rbase = wave * 16;  // this wave's 16 text rows

    const uint32_t ldsAddr0 = (uint32_t)(uintptr_t)&ldsV[0][tid * LDS_ROW_U16];
    const uint32_t ldsAddr1 = (uint32_t)(uintptr_t)&ldsV[1][tid * LDS_ROW_U16];
    const unsigned short* grow = Vbase + (size_t)tid * DD;   // this thread's V row

    // Prologue: async-stage slab kb=0 into buffer 0
    stage_async(ldsAddr0, (const void*)(grow + 0));

    int cur = 0;
    for (int kb = 0; kb < DD; kb += 32) {
        // A fragment (16x32 f16, contiguous k = 16*sel + s) from global; latency overlaps prior compute
        Frag a;
        const f4* arow = (const f4*)(Tbase + (size_t)(rbase + lane_lo) * DD + kb);
        a.f[0] = arow[2 * sel];       // K = kb + sel*16 .. +7
        a.f[1] = arow[2 * sel + 1];   // K = kb + sel*16 + 8 .. +15

        if (kb + 32 < DD) {
            // Kick off next slab into the other buffer (trailing barrier of previous
            // iteration guarantees all waves are done reading it).
            stage_async(cur ? ldsAddr0 : ldsAddr1, (const void*)(grow + kb + 32));
            asm volatile("s_wait_asynccnt 0x4" ::: "memory");  // current slab landed
        } else {
            asm volatile("s_wait_asynccnt 0x0" ::: "memory");
        }
        __syncthreads();          // all waves' slab writes visible

        const unsigned short* lv = ldsV[cur];
        // 2-deep software pipeline with two live fragment register octets.
        Frag b0, b1;
        loadB(b0, lv, 0, lane_lo, sel);
        #pragma unroll
        for (int t = 0; t < 16; t += 2) {
            loadB(b1, lv, t + 1, lane_lo, sel);
            acc[t] = __builtin_amdgcn_wmma_f32_16x16x32_f16(
                false, a.h, false, b0.h, (short)0, acc[t], false, false);
            if (t + 2 < 16) loadB(b0, lv, t + 2, lane_lo, sel);
            acc[t + 1] = __builtin_amdgcn_wmma_f32_16x16x32_f16(
                false, a.h, false, b1.h, (short)0, acc[t + 1], false, false);
        }
        // Explicit scheduling pipeline: DS-read group for fragment t+1 must schedule
        // before WMMA t, keeping one fragment in flight under each WMMA:
        //   F0, F1, W0, F2, W1, ..., F15, W14, W15
        __builtin_amdgcn_sched_group_barrier(SG_DSREAD, 2, 0);   // F0
        #pragma unroll
        for (int t = 0; t < 15; ++t) {
            __builtin_amdgcn_sched_group_barrier(SG_DSREAD, 2, 0); // F(t+1)
            __builtin_amdgcn_sched_group_barrier(SG_WMMA,   1, 0); // W(t)
        }
        __builtin_amdgcn_sched_group_barrier(SG_WMMA, 1, 0);     // W15

        __syncthreads();          // done reading buf[cur]; it may be overwritten next iter
        cur ^= 1;
    }

    // ---- Fused epilogue ----
    // C layout: VGPR c, lanes 0-15 -> (M=c, N=lane), lanes 16-31 -> (M=c+8, N=lane-16)
    float regsum = 0.f;
    float rmax[8];
    #pragma unroll
    for (int c = 0; c < 8; ++c) rmax[c] = -1e30f;
    #pragma unroll
    for (int t = 0; t < 16; ++t) {
        #pragma unroll
        for (int c = 0; c < 8; ++c) {
            float s = acc[t][c];
            rmax[c] = fmaxf(rmax[c], s);
            float cn = fmaxf(fminf(s, 0.f), -20.f);
            regsum += cn * cn;
        }
    }
    // cross-lane max within each 16-lane half (xor masks < 16 keep halves separate)
    #pragma unroll
    for (int c = 0; c < 8; ++c) {
        float m = rmax[c];
        #pragma unroll
        for (int d = 1; d < 16; d <<= 1) m = fmaxf(m, __shfl_xor(m, d, 32));
        rmax[c] = m;
    }
    // masked sum of row maxima; row = rbase + sel*8 + c
    float part = 0.f;
    if (lane_lo == 0) {
        #pragma unroll
        for (int c = 0; c < 8; ++c) {
            int r = rbase + sel * 8 + c;
            part += rmax[c] * (float)amask[i * NT + r];
        }
    }
    part += __shfl_xor(part, 16, 32);   // lane 0 now holds both halves

    #pragma unroll
    for (int d = 1; d < 32; d <<= 1) regsum += __shfl_xor(regsum, d, 32);

    if (lane == 0) { ldsRed[wave] = part; ldsRed[8 + wave] = regsum; }
    __syncthreads();
    if (tid == 0) {
        float ms = 0.f, rs = 0.f;
        #pragma unroll
        for (int w = 0; w < 8; ++w) { ms += ldsRed[w]; rs += ldsRed[8 + w]; }
        msum[i * BB + j] = ms;
        regp[i * BB + j] = rs;
    }
}

// ---------------- Kernel 3: InfoNCE on 64x64 + regularization ----------------
__global__ __launch_bounds__(256) void loss_kernel(
    const int* __restrict__ amask,
    const float* __restrict__ msum,
    const float* __restrict__ regp,
    float* __restrict__ out)
{
    __shared__ float clip[BB * BB];  // 16 KB
    __shared__ float cnt[BB];
    __shared__ float lossArr[BB];
    __shared__ float red[256];

    const int tid = threadIdx.x;
    if (tid < BB) {
        int s = 0;
        for (int r = 0; r < NT; ++r) s += amask[tid * NT + r];
        float c = (float)s;
        cnt[tid] = (c < EPSF) ? EPSF : c;
    }
    __syncthreads();
    for (int e = tid; e < BB * BB; e += 256) clip[e] = msum[e] / cnt[e >> 6];
    float rs = 0.f;
    for (int e = tid; e < BB * BB; e += 256) rs += regp[e];
    red[tid] = rs;
    __syncthreads();

    if (tid < BB) {
        const int i = tid;
        float m = -1e30f;
        for (int jj = 0; jj < BB; ++jj) m = fmaxf(m, clip[i * BB + jj]);
        float se = 0.f;
        for (int jj = 0; jj < BB; ++jj) se += expf(clip[i * BB + jj] - m);
        float lt = -(clip[i * BB + i] - (m + logf(se)));

        float m2 = -1e30f;
        for (int jj = 0; jj < BB; ++jj) m2 = fmaxf(m2, clip[jj * BB + i]);
        float se2 = 0.f;
        for (int jj = 0; jj < BB; ++jj) se2 += expf(clip[jj * BB + i] - m2);
        float lv = -(clip[i * BB + i] - (m2 + logf(se2)));
        lossArr[i] = lt + lv;
    }
    __syncthreads();
    if (tid == 0) {
        float ls = 0.f;
        for (int ii = 0; ii < BB; ++ii) ls += lossArr[ii];
        float contrastive = ls / (2.f * (float)BB);
        float rtot = 0.f;
        for (int t = 0; t < 256; ++t) rtot += red[t];
        float reg = 0.15f * (rtot / (float)((long long)BB * BB * NT * NV));
        out[0] = contrastive + reg;
    }
}

extern "C" void kernel_launch(void* const* d_in, const int* in_sizes, int n_in,
                              void* d_out, int out_size, void* d_ws, size_t ws_size,
                              hipStream_t stream) {
    const float* text = (const float*)d_in[0];   // (64,128,512) f32
    const float* vis  = (const float*)d_in[1];   // (64,256,512) f32
    const int*   amask = (const int*)d_in[2];    // (64,128) i32

    char* ws = (char*)d_ws;
    unsigned short* tn = (unsigned short*)ws;                         // 8192*512 f16  = 8 MB
    unsigned short* vn = (unsigned short*)(ws + 8388608);             // 16384*512 f16 = 16 MB
    float* msum = (float*)(ws + 8388608 + 16777216);                  // 4096 f32
    float* regp = (float*)(ws + 8388608 + 16777216 + 16384);          // 4096 f32

    normalize_f16_kernel<<<dim3(3072), dim3(256), 0, stream>>>(text, vis, tn, vn);
    pair_gemm_kernel<<<dim3(64, 64), dim3(256), 0, stream>>>(tn, vn, amask, msum, regp);
    loss_kernel<<<dim3(1), dim3(256), 0, stream>>>(amask, msum, regp, (float*)d_out);
}
